// CharLSTM_78151224918718
// MI455X (gfx1250) — matrix operand order repaired and checked by reading, not compiled, not run on hardware
//
#include <hip/hip_runtime.h>
#include <hip/hip_bf16.h>

// ---------------------------------------------------------------------------
// Bidirectional char-LSTM for MI455X (gfx1250, wave32, WMMA).
//
//  - One workgroup (256 thr = 8 wave32) owns a tile of 32 words and runs the
//    entire 20-step recurrence on-chip.  A = [x_t | h] (32x384 f16) staged in
//    LDS; gates = A @ Wc^T via v_wmma_f32_16x16x32_f16 (K=384 -> 12 k-tiles).
//  - Two 16-row A tiles share every B fragment -> 2 WMMAs per L2 weight load;
//    B fragments are software-pipelined one tile ahead so each load has two
//    WMMAs of issue distance before its wait.
//  - Wc is the band-permuted [Wih|Whh] (f16, 1024x384) so each wave's 128
//    output columns carry i/f/g/o for "its" 32 hidden units -> the LSTM cell
//    update is lane-local on the accumulator fragments.
//  - Branch-free activations on v_exp_f32/v_rcp_f32 (no libm tanh exec-mask
//    gymnastics between barriers).
//  - x is streamed with non-temporal loads (and out with NT stores) so the
//    192MB L2 stays dedicated to the 1.5MB of packed weights.
// ---------------------------------------------------------------------------

typedef _Float16 v16h __attribute__((ext_vector_type(16)));
typedef _Float16 v8h  __attribute__((ext_vector_type(8)));
typedef float    v8f  __attribute__((ext_vector_type(8)));

#define NWORDS   32768
#define TMAX     20
#define NCHARS   128
#define HID      256
#define KDIM     (NCHARS + HID)    // 384
#define NGATES   (4 * HID)         // 1024
#define TILE_B   32                // words per workgroup (2 x 16-row WMMA tiles)
#define LDA      392               // padded f16 row length (784B = 49*16B)

__device__ __forceinline__ float fast_sigmoid(float x) {
    return __builtin_amdgcn_rcpf(1.0f + __expf(-x));
}
__device__ __forceinline__ float fast_tanh(float x) {
    // 1 - 2/(1+e^{2x}): exact limits +-1, branch-free, 2 trans ops
    return 1.0f - 2.0f * __builtin_amdgcn_rcpf(1.0f + __expf(2.0f * x));
}

// --------------------------- weight pre-pack -------------------------------
// New row index n = band*128 + gate*32 + unit  (band=hu>>5, unit=hu&31)
// maps to original gate row  orig = gate*256 + band*32 + unit.
// Columns: [0,128) = Wih row, [128,384) = Whh row.  Also bsum = bih+bhh.
__global__ __launch_bounds__(128)
void lstm_pack_weights(const float* __restrict__ Wih,
                       const float* __restrict__ Whh,
                       const float* __restrict__ bih,
                       const float* __restrict__ bhh,
                       _Float16* __restrict__ Wc,
                       float* __restrict__ bsum)
{
    const int n_new = blockIdx.x;            // 0..1023
    const int band  = n_new >> 7;
    const int gate  = (n_new >> 5) & 3;
    const int unit  = n_new & 31;
    const int orig  = gate * HID + band * 32 + unit;

    for (int k = threadIdx.x; k < KDIM; k += blockDim.x) {
        float w = (k < NCHARS) ? Wih[(size_t)orig * NCHARS + k]
                               : Whh[(size_t)orig * HID + (k - NCHARS)];
        Wc[(size_t)n_new * KDIM + k] = (_Float16)w;
    }
    if (threadIdx.x == 0) bsum[n_new] = bih[orig] + bhh[orig];
}

// ------------------------------ main kernel --------------------------------
__global__ __launch_bounds__(256)
void lstm_tile_kernel(const float* __restrict__ x,
                      const int*   __restrict__ len,
                      const _Float16* __restrict__ Wc,
                      const float* __restrict__ bsum,
                      float* __restrict__ out,
                      int reverse)
{
    __shared__ _Float16 Alds[TILE_B * LDA];     // [32][392]: cols 0..127 = x_t, 128..383 = h (f16)
    __shared__ float    h32[TILE_B * HID];      // fp32 copy of hidden state
    __shared__ int      lenS[TILE_B];

    const int tid   = threadIdx.x;
    const int wave  = tid >> 5;                 // 0..7  (wave32)
    const int lane  = tid & 31;
    const int hf    = lane >> 4;                // lane half (0/1)
    const int col   = lane & 15;                // N-column / A-row within tile
    const int word0 = blockIdx.x * TILE_B;
    const int nbase = wave * 128;               // this wave's gate-column base

    // per-lane B row pointer base: n = nbase + a*16 + col, k-offset hf*16
    const _Float16* __restrict__ wrow =
        Wc + (size_t)(nbase + col) * KDIM + hf * 16;

    if (tid < TILE_B) lenS[tid] = len[word0 + tid];
    for (int i = tid; i < TILE_B * HID; i += blockDim.x) {
        h32[i] = 0.0f;
        const int r = i >> 8, c = i & 255;
        Alds[r * LDA + NCHARS + c] = (_Float16)0.0f;
    }

    // per-lane persistent cell state:
    // cst[m][sub*8+v] for word-row m*16 + 8*hf+v, hidden unit wave*32+16*sub+col
    float cst[2][16];
#pragma unroll
    for (int m = 0; m < 2; ++m)
#pragma unroll
        for (int i = 0; i < 16; ++i) cst[m][i] = 0.0f;

    // lane-local biases (band-permuted layout)
    float bI[2], bF[2], bG[2], bO[2];
#pragma unroll
    for (int sub = 0; sub < 2; ++sub) {
        const int hu = 16 * sub + col;
        bI[sub] = bsum[nbase +  0 + hu];
        bF[sub] = bsum[nbase + 32 + hu];
        bG[sub] = bsum[nbase + 64 + hu];
        bO[sub] = bsum[nbase + 96 + hu];
    }

    for (int s = 0; s < TMAX; ++s) {
        const int t = reverse ? (TMAX - 1 - s) : s;

        __syncthreads();                        // prior GEMM done before x overwrite
        // stage x_t tile (32x128 f32 -> f16); NT loads keep Wc hot in L2
        for (int i = tid; i < TILE_B * NCHARS; i += blockDim.x) {
            const int r = i >> 7, c = i & 127;
            const float v = __builtin_nontemporal_load(
                &x[((size_t)(word0 + r) * TMAX + t) * NCHARS + c]);
            Alds[r * LDA + c] = (_Float16)v;
        }
        __syncthreads();

        // ---- gates[32][nbase..nbase+127] = A(32x384) @ Wc^T slice ----
        v8f acc[2][8] = {};
        for (int kt = 0; kt < KDIM / 32; ++kt) {
            const int k0 = kt * 32;
            // A fragments: lane = row `col`; half 0 -> K{0..15}, half 1 -> K{16..31}
            v16h afr[2];
#pragma unroll
            for (int m = 0; m < 2; ++m) {
                const int arow = m * 16 + col;
                const v8h lo = *(const v8h*)&Alds[arow * LDA + k0 + hf * 16];
                const v8h hi = *(const v8h*)&Alds[arow * LDA + k0 + hf * 16 + 8];
#pragma unroll
                for (int i = 0; i < 8; ++i) { afr[m][i] = lo[i]; afr[m][8 + i] = hi[i]; }
            }
            // B fragments, software-pipelined one N-tile ahead:
            // lane = column n; lanes 0-15 hold K 0..15, 16-31 hold K 16..31
            v16h bcur = *(const v16h*)(wrow + k0);
#pragma unroll
            for (int a = 0; a < 8; ++a) {
                v16h bnext;
                if (a < 7)
                    bnext = *(const v16h*)(wrow + (size_t)(a + 1) * 16 * KDIM + k0);
                acc[0][a] = __builtin_amdgcn_wmma_f32_16x16x32_f16(
                                false, afr[0], false, bcur, (short)0, acc[0][a], false, false);
                acc[1][a] = __builtin_amdgcn_wmma_f32_16x16x32_f16(
                                false, afr[1], false, bcur, (short)0, acc[1][a], false, false);
                if (a < 7) bcur = bnext;
            }
        }
        __syncthreads();                        // all waves done reading h before update

        // ---- lane-local LSTM cell update (branch-free activations) ----
#pragma unroll
        for (int m = 0; m < 2; ++m) {
#pragma unroll
            for (int sub = 0; sub < 2; ++sub) {
                const int hu = wave * 32 + 16 * sub + col;   // global hidden unit
#pragma unroll
                for (int v = 0; v < 8; ++v) {
                    const int  row = m * 16 + 8 * hf + v;    // word row in tile
                    const bool act = (t < lenS[row]);
                    const float iv = fast_sigmoid(acc[m][0 + sub][v] + bI[sub]);
                    const float fv = fast_sigmoid(acc[m][2 + sub][v] + bF[sub]);
                    const float gv = fast_tanh   (acc[m][4 + sub][v] + bG[sub]);
                    const float ov = fast_sigmoid(acc[m][6 + sub][v] + bO[sub]);
                    const float cn = fv * cst[m][sub * 8 + v] + iv * gv;
                    const float hn = ov * fast_tanh(cn);
                    if (act) {
                        cst[m][sub * 8 + v] = cn;
                        h32[row * HID + hu] = hn;
                        Alds[row * LDA + NCHARS + hu] = (_Float16)hn;
                    }
                }
            }
        }
    }

    __syncthreads();
    // final hidden -> out[word][reverse*256 + c]   (concat [h_fwd, h_bwd]); NT stores
    for (int i = tid; i < TILE_B * HID; i += blockDim.x) {
        const int r = i >> 8, c = i & 255;
        __builtin_nontemporal_store(
            h32[i], &out[(size_t)(word0 + r) * (2 * HID) + reverse * HID + c]);
    }
}

// ------------------------------ launcher -----------------------------------
extern "C" void kernel_launch(void* const* d_in, const int* in_sizes, int n_in,
                              void* d_out, int out_size, void* d_ws, size_t ws_size,
                              hipStream_t stream)
{
    const float* x     = (const float*)d_in[0];
    const int*   lens  = (const int*)  d_in[1];
    const float* Wih_f = (const float*)d_in[2];
    const float* Whh_f = (const float*)d_in[3];
    const float* bih_f = (const float*)d_in[4];
    const float* bhh_f = (const float*)d_in[5];
    const float* Wih_b = (const float*)d_in[6];
    const float* Whh_b = (const float*)d_in[7];
    const float* bih_b = (const float*)d_in[8];
    const float* bhh_b = (const float*)d_in[9];
    float* out = (float*)d_out;

    // workspace layout
    char* ws = (char*)d_ws;
    const size_t wcBytes = (size_t)NGATES * KDIM * sizeof(_Float16);  // 768 KB
    _Float16* Wc_f   = (_Float16*)(ws);
    _Float16* Wc_b   = (_Float16*)(ws + wcBytes);
    float*    bsum_f = (float*)   (ws + 2 * wcBytes);
    float*    bsum_b = (float*)   (ws + 2 * wcBytes + NGATES * sizeof(float));

    hipLaunchKernelGGL(lstm_pack_weights, dim3(NGATES), dim3(128), 0, stream,
                       Wih_f, Whh_f, bih_f, bhh_f, Wc_f, bsum_f);
    hipLaunchKernelGGL(lstm_pack_weights, dim3(NGATES), dim3(128), 0, stream,
                       Wih_b, Whh_b, bih_b, bhh_b, Wc_b, bsum_b);

    const int nTiles = NWORDS / TILE_B;   // 1024
    hipLaunchKernelGGL(lstm_tile_kernel, dim3(nTiles), dim3(256), 0, stream,
                       x, lens, Wc_f, bsum_f, out, 0);
    hipLaunchKernelGGL(lstm_tile_kernel, dim3(nTiles), dim3(256), 0, stream,
                       x, lens, Wc_b, bsum_b, out, 1);
}
